// NormalizingFlow_40218073759777
// MI455X (gfx1250) — matrix operand order repaired and checked by reading, not compiled, not run on hardware
//
#include <hip/hip_runtime.h>

#define LNUM 8
#define DDIM 8
#define KBIN 8
#define HDIM 128
#define POUT 200
#define POUT_PAD 208
#define RMIN_F (-5.0f)
#define RMAX_F (5.0f)
#define MIN_BIN_F 0.001f
#define MIN_SLOPE_F 0.001f
#define SOFTPLUS_OFFS 0.53974175f   /* log(expm1(1 - MIN_SLOPE)) */

typedef __attribute__((ext_vector_type(16))) __bf16 v16bf;
typedef __attribute__((ext_vector_type(8)))  __bf16 v8bf;
typedef __attribute__((ext_vector_type(8)))  float  v8f;

// ---------------- fragment id helpers (packed bf16 B-fragments in d_ws) ----
__host__ __device__ inline int frag_win (int l, int nt)                { return l*8 + nt; }
__host__ __device__ inline int frag_whid(int l, int s, int nt, int kc) { return 64 + l*64 + s*32 + nt*4 + kc; }
__host__ __device__ inline int frag_wout(int l, int nt, int kc)        { return 576 + l*52 + nt*4 + kc; }

#define NFRAGS     992
#define FRAG_ELEMS 512                      /* 32 lanes * 16 bf16 */
#define FRAG_BYTES (NFRAGS * FRAG_ELEMS * 2)

// ---------------- per-wave LDS slice layout (bytes) ------------------------
#define XT_OFF   0        /* 16*8  f32  =   512 */
#define HA_OFF   512      /* 16*128 bf16 = 4096 */
#define HB_OFF   4608     /* 16*128 bf16 = 4096 */
#define PB_OFF   8704     /* 16*208 f32 = 13312 */
#define LDA_OFF  22016    /* 16 f32 */
#define LDP_OFF  22080    /* 32 f32 */
#define WAVE_BYTES 22272
#define WAVES_PER_BLOCK 2
#define SMEM_BYTES (WAVES_PER_BLOCK * WAVE_BYTES)

// ============================================================================
// Prep kernel 1: pack weights as WMMA bf16 B-fragments.
// B k order packed here (contiguous, same as A): lane<16 -> N=lane, elems j=0..15 are K=kc*32+j;
//                           lane>=16 -> N=lane-16, elems j are K=kc*32+16+j.
// Each lane's 16 bf16 stored contiguously (32B) for a single vector load.
// Mask is folded into w_in; w_out padded to 208 output rows with zeros.
// ============================================================================
__global__ void nf_pack_weights(const float* __restrict__ w_in,
                                const float* __restrict__ w_hid,
                                const float* __restrict__ w_out,
                                const float* __restrict__ masks,
                                __bf16* __restrict__ wpk)
{
    const int f    = blockIdx.x;
    const int lane = threadIdx.x;       // 32 threads
    const int n16  = lane & 15;
    const int kb16 = (lane >= 16) ? 16 : 0;

    float vals[16];
    if (f < 64) {                       // w_in : (L,128,8), K padded 8->32
        const int l = f >> 3, nt = f & 7;
        const int n = nt * 16 + n16;
        #pragma unroll
        for (int j = 0; j < 16; ++j) {
            const int k = kb16 + j;
            float v = 0.f;
            if (k < DDIM) v = w_in[(l * HDIM + n) * DDIM + k] * masks[l * DDIM + k];
            vals[j] = v;
        }
    } else if (f < 576) {               // w_hid : (L,2,128,128)
        const int idx = f - 64;
        const int l = idx >> 6, rem = idx & 63;
        const int s = rem >> 5, rem2 = rem & 31;
        const int nt = rem2 >> 2, kc = rem2 & 3;
        const int n = nt * 16 + n16;
        const int kb = kc * 32 + kb16;
        #pragma unroll
        for (int j = 0; j < 16; ++j)
            vals[j] = w_hid[(((l * 2 + s) * HDIM) + n) * HDIM + kb + j];
    } else {                            // w_out : (L,200,128), N padded 200->208
        const int idx = f - 576;
        const int l = idx / 52, rem = idx % 52;
        const int nt = rem >> 2, kc = rem & 3;
        const int n = nt * 16 + n16;
        const int kb = kc * 32 + kb16;
        #pragma unroll
        for (int j = 0; j < 16; ++j) {
            float v = 0.f;
            if (n < POUT) v = w_out[(l * POUT + n) * HDIM + kb + j];
            vals[j] = v;
        }
    }
    __bf16* dst = wpk + (size_t)f * FRAG_ELEMS + lane * 16;
    #pragma unroll
    for (int j = 0; j < 16; ++j) dst[j] = (__bf16)vals[j];
}

// ============================================================================
// Prep kernel 2: PLU -> permuted 8x8 W per layer + sum(log diag).
// ============================================================================
__global__ void nf_plu_prep(const float* __restrict__ pl,
                            const float* __restrict__ pu,
                            const float* __restrict__ plog,
                            const int*   __restrict__ pidx,
                            float* __restrict__ plw,
                            float* __restrict__ plsum)
{
    const int l = blockIdx.x;
    const int t = threadIdx.x;          // 64 threads
    __shared__ float W0[8][8];
    const int i = t >> 3, j = t & 7;
    float s = 0.f;
    #pragma unroll
    for (int m = 0; m < 8; ++m) {
        const float lm = (m < i) ? pl[(l * 8 + i) * 8 + m] : ((m == i) ? 1.f : 0.f);
        const float um = (m < j) ? pu[(l * 8 + m) * 8 + j]
                                 : ((m == j) ? expf(plog[l * 8 + m]) : 0.f);
        s += lm * um;
    }
    W0[i][j] = s;
    __syncthreads();
    const int pi = pidx[l * 8 + i];
    plw[l * 64 + i * 8 + j] = W0[pi][j];
    if (t == 0) {
        float ps = 0.f;
        #pragma unroll
        for (int d = 0; d < 8; ++d) ps += plog[l * 8 + d];
        plsum[l] = ps;
    }
}

// ============================================================================
// Device helpers — branchless, native-transcendental (v_exp_f32 / v_log_f32)
// ============================================================================
// gelu(x) = 0.5x(1+tanh(u)) = x * sigmoid(2u),  u = sqrt(2/pi)(x+0.044715x^3)
__device__ inline float gelu_f(float x) {
    const float c = 0.7978845608028654f;
    const float u2 = 2.f * c * (x + 0.044715f * x * x * x);
    return x / (1.f + __expf(-u2));      // overflow-safe: x/inf -> 0
}

__device__ inline float softplus_f(float t) {   // stable, branchless
    return fmaxf(t, 0.f) + __logf(1.f + __expf(-fabsf(t)));
}

__device__ inline v8f vzero8() {
    v8f z = {0.f, 0.f, 0.f, 0.f, 0.f, 0.f, 0.f, 0.f};
    return z;
}

// A k order used here (contiguous, same as B packing): lane<16 -> M=lane, elems 0..15 = K base+0..15,
// where base = kc*32 + (lane>=16)*16.
__device__ inline v16bf load_afrag(const __bf16* __restrict__ h, int lane, int kc) {
    const int m = lane & 15;
    const int lhalf = lane >> 4;
    const __bf16* row = h + m * HDIM + kc * 32 + lhalf * 16;
    union { v16bf v; struct { v8bf lo; v8bf hi; } s; } u;
    u.s.lo = *(const v8bf*)(row);
    u.s.hi = *(const v8bf*)(row + 8);
    return u.v;
}

__device__ inline v16bf load_bfrag(const __bf16* __restrict__ wpk, int frag, int lane) {
    return *(const v16bf*)(wpk + (size_t)frag * FRAG_ELEMS + lane * 16);
}

// Rational-quadratic spline forward for one scalar; fully unrolled (no scratch).
__device__ inline void rqs_fwd(float xv, const float* __restrict__ p,
                               float& yo, float& lo)
{
    const float span  = RMAX_F - RMIN_F;
    const float scale = span - KBIN * MIN_BIN_F;
    float ew[8], eh[8], wj[8], hj[8], sl[9], kx[9], ky[9];

    float mw = p[0], mh = p[8];
    #pragma unroll
    for (int j = 1; j < 8; ++j) { mw = fmaxf(mw, p[j]); mh = fmaxf(mh, p[8 + j]); }
    float sw = 0.f, sh = 0.f;
    #pragma unroll
    for (int j = 0; j < 8; ++j) {
        ew[j] = __expf(p[j] - mw);       sw += ew[j];   // arg <= 0: safe
        eh[j] = __expf(p[8 + j] - mh);   sh += eh[j];
    }
    const float isw = scale / sw, ish = scale / sh;
    #pragma unroll
    for (int j = 0; j < 8; ++j) {
        wj[j] = MIN_BIN_F + ew[j] * isw;
        hj[j] = MIN_BIN_F + eh[j] * ish;
    }
    #pragma unroll
    for (int j = 0; j < 9; ++j)
        sl[j] = MIN_SLOPE_F + softplus_f(p[16 + j] + SOFTPLUS_OFFS);

    kx[0] = RMIN_F; ky[0] = RMIN_F;
    #pragma unroll
    for (int j = 0; j < 8; ++j) { kx[j + 1] = kx[j] + wj[j]; ky[j + 1] = ky[j] + hj[j]; }

    const bool inside = (xv > RMIN_F) && (xv < RMAX_F);
    const float xc = fminf(fmaxf(xv, RMIN_F), RMAX_F);
    int bin = 0;
    #pragma unroll
    for (int j = 1; j <= 8; ++j) bin += (xc >= kx[j]) ? 1 : 0;
    bin = (bin > 7) ? 7 : bin;

    float xk = 0.f, yk = 0.f, wk = 1.f, hk = 1.f, dk = 1.f, dk1 = 1.f;
    #pragma unroll
    for (int j = 0; j < 8; ++j) {
        const bool sel = (bin == j);
        xk  = sel ? kx[j]     : xk;
        yk  = sel ? ky[j]     : yk;
        wk  = sel ? wj[j]     : wk;
        hk  = sel ? hj[j]     : hk;
        dk  = sel ? sl[j]     : dk;
        dk1 = sel ? sl[j + 1] : dk1;
    }
    const float sk   = hk / wk;
    const float xi   = (xc - xk) / wk;
    const float omx  = 1.f - xi;
    const float z    = xi * omx;
    const float den  = sk + (dk1 + dk - 2.f * sk) * z;
    const float iden = 1.f / den;
    const float num  = dk1 * xi * xi + 2.f * sk * z + dk * omx * omx;
    const float y    = yk + hk * (sk * xi * xi + dk * z) * iden;
    // 2log(sk)+log(num)-2log(den) folded into a single native log:
    const float ld   = __logf(sk * sk * num * iden * iden);
    yo = inside ? y  : xv;
    lo = inside ? ld : 0.f;
}

// ============================================================================
// Main kernel: each wave32 owns one 16-row tile through all 8 layers.
// No inter-wave sharing -> no barriers; LDS ops are in-order within a wave.
// ============================================================================
__global__ void __launch_bounds__(64)
nf_flow_main(const float* __restrict__ xg,   const float* __restrict__ masks,
             const float* __restrict__ b_in, const float* __restrict__ b_hid,
             const float* __restrict__ b_out,const float* __restrict__ plu_b,
             const __bf16* __restrict__ wpk, const float* __restrict__ plw,
             const float* __restrict__ plsum, float* __restrict__ out, int Bn)
{
    extern __shared__ char smem[];
    const int lane = threadIdx.x & 31;
    const int wave = threadIdx.x >> 5;
    char* wbase = smem + wave * WAVE_BYTES;
    float*  xt     = (float*)(wbase + XT_OFF);
    __bf16* hA     = (__bf16*)(wbase + HA_OFF);
    __bf16* hB     = (__bf16*)(wbase + HB_OFF);
    float*  pbuf   = (float*)(wbase + PB_OFF);
    float*  ldacc  = (float*)(wbase + LDA_OFF);
    float*  ldpart = (float*)(wbase + LDP_OFF);

    const int lhalf = lane >> 4;
    const int l15   = lane & 15;
    const int numTiles = Bn >> 4;
    const int nWaves = gridDim.x * WAVES_PER_BLOCK;

    for (int tile = blockIdx.x * WAVES_PER_BLOCK + wave; tile < numTiles; tile += nWaves) {
        if (lane < 16) {
            const float* src = xg + (size_t)(tile * 16 + lane) * DDIM;
            float4 a0 = ((const float4*)src)[0];
            float4 a1 = ((const float4*)src)[1];
            ((float4*)(xt + lane * 8))[0] = a0;
            ((float4*)(xt + lane * 8))[1] = a1;
            ldacc[lane] = 0.f;
        }

        for (int l = 0; l < LNUM; ++l) {
            // ---- input GEMM: (16x8 x, zero-padded to K=32) @ (w_in*mask)^T ----
            v16bf aIn;
            #pragma unroll
            for (int j = 0; j < 16; ++j) aIn[j] = (__bf16)0.0f;
            if (lhalf == 0) {
                #pragma unroll
                for (int j = 0; j < 8; ++j) aIn[j] = (__bf16)xt[l15 * 8 + j];
            }
            #pragma unroll
            for (int nt = 0; nt < 8; ++nt) {
                v8f acc = vzero8();
                v16bf bf = load_bfrag(wpk, frag_win(l, nt), lane);
                acc = __builtin_amdgcn_wmma_f32_16x16x32_bf16(
                        false, aIn, false, bf, (short)0, acc, false, false);
                const int n = nt * 16 + l15;
                const float bias = b_in[l * HDIM + n];
                #pragma unroll
                for (int v = 0; v < 8; ++v)
                    hA[(lhalf * 8 + v) * HDIM + n] = (__bf16)gelu_f(acc[v] + bias);
            }

            // ---- hidden GEMMs: 2 x (16x128 @ 128x128), ping-pong LDS -------
            __bf16* srcb = hA; __bf16* dstb = hB;
            #pragma unroll
            for (int s = 0; s < 2; ++s) {
                #pragma unroll
                for (int nt = 0; nt < 8; ++nt) {
                    v8f acc = vzero8();
                    #pragma unroll
                    for (int kc = 0; kc < 4; ++kc) {
                        v16bf a  = load_afrag(srcb, lane, kc);
                        v16bf bf = load_bfrag(wpk, frag_whid(l, s, nt, kc), lane);
                        acc = __builtin_amdgcn_wmma_f32_16x16x32_bf16(
                                false, a, false, bf, (short)0, acc, false, false);
                    }
                    const int n = nt * 16 + l15;
                    const float bias = b_hid[(l * 2 + s) * HDIM + n];
                    #pragma unroll
                    for (int v = 0; v < 8; ++v)
                        dstb[(lhalf * 8 + v) * HDIM + n] = (__bf16)gelu_f(acc[v] + bias);
                }
                __bf16* t = srcb; srcb = dstb; dstb = t;
            }

            // ---- output GEMM: 16x128 @ 128x208(padded) -> pbuf (f32) -------
            #pragma unroll
            for (int nt = 0; nt < 13; ++nt) {
                v8f acc = vzero8();
                #pragma unroll
                for (int kc = 0; kc < 4; ++kc) {
                    v16bf a  = load_afrag(srcb, lane, kc);
                    v16bf bf = load_bfrag(wpk, frag_wout(l, nt, kc), lane);
                    acc = __builtin_amdgcn_wmma_f32_16x16x32_bf16(
                            false, a, false, bf, (short)0, acc, false, false);
                }
                const int n = nt * 16 + l15;
                const float bias = (n < POUT) ? b_out[l * POUT + n] : 0.f;
                #pragma unroll
                for (int v = 0; v < 8; ++v)
                    pbuf[(lhalf * 8 + v) * POUT_PAD + n] = acc[v] + bias;
            }

            // ---- RQS spline: 4 (row,dim) pairs per lane --------------------
            {
                const int r = lane >> 1;
                const int dbase = (lane & 1) * 4;
                float ljsum = 0.f;
                #pragma unroll
                for (int dd = 0; dd < 4; ++dd) {
                    const int d = dbase + dd;
                    const float xv = xt[r * 8 + d];
                    const float mk = masks[l * 8 + d];
                    float yv, lj;
                    rqs_fwd(xv, pbuf + r * POUT_PAD + d * 25, yv, lj);
                    const float y = mk * xv + (1.f - mk) * yv;
                    pbuf[r * POUT_PAD + POUT + d] = y;   // stash y in pad cols
                    ljsum += (1.f - mk) * lj;
                }
                ldpart[lane] = ljsum;
            }

            // ---- PLU 8x8 matvec per row + logdet accumulation --------------
            if (lane < 16) {
                const int r = lane;
                float yrow[8];
                #pragma unroll
                for (int j = 0; j < 8; ++j) yrow[j] = pbuf[r * POUT_PAD + POUT + j];
                const float* W = plw + l * 64;
                #pragma unroll
                for (int i = 0; i < 8; ++i) {
                    float s2 = plu_b[l * 8 + i];
                    #pragma unroll
                    for (int j = 0; j < 8; ++j) s2 += W[i * 8 + j] * yrow[j];
                    xt[r * 8 + i] = s2;
                }
                ldacc[r] += ldpart[2 * r] + ldpart[2 * r + 1] + plsum[l];
            }
        }

        // ---- write outputs: x_final (B,8) then logdet (B,) -----------------
        if (lane < 16) {
            const int row = tile * 16 + lane;
            float* dst = out + (size_t)row * 8;
            const float* srow = xt + lane * 8;
            ((float4*)dst)[0] = ((const float4*)srow)[0];
            ((float4*)dst)[1] = ((const float4*)srow)[1];
            out[(size_t)Bn * 8 + row] = ldacc[lane];
        }
    }
}

// ============================================================================
extern "C" void kernel_launch(void* const* d_in, const int* in_sizes, int n_in,
                              void* d_out, int out_size, void* d_ws, size_t ws_size,
                              hipStream_t stream)
{
    const float* x      = (const float*)d_in[0];
    const float* masks  = (const float*)d_in[1];
    const float* w_in   = (const float*)d_in[2];
    const float* b_in   = (const float*)d_in[3];
    const float* w_hid  = (const float*)d_in[4];
    const float* b_hid  = (const float*)d_in[5];
    const float* w_out  = (const float*)d_in[6];
    const float* b_out  = (const float*)d_in[7];
    const float* pl     = (const float*)d_in[8];
    const float* pu     = (const float*)d_in[9];
    const float* plog   = (const float*)d_in[10];
    const float* pb     = (const float*)d_in[11];
    const int*   pidx   = (const int*)d_in[12];

    const int Bn = in_sizes[0] / DDIM;

    __bf16* wpk  = (__bf16*)d_ws;
    float*  plw  = (float*)((char*)d_ws + FRAG_BYTES);
    float*  plsum = plw + LNUM * 64;

    nf_pack_weights<<<NFRAGS, 32, 0, stream>>>(w_in, w_hid, w_out, masks, wpk);
    nf_plu_prep<<<LNUM, 64, 0, stream>>>(pl, pu, plog, pidx, plw, plsum);

    const int numTiles = Bn >> 4;
    int blocks = 4096;
    if (blocks * WAVES_PER_BLOCK > numTiles)
        blocks = (numTiles + WAVES_PER_BLOCK - 1) / WAVES_PER_BLOCK;
    nf_flow_main<<<dim3(blocks), dim3(64), SMEM_BYTES, stream>>>(
        x, masks, b_in, b_hid, b_out, pb, wpk, plw, plsum, (float*)d_out, Bn);
}
